// GraphAttentionLayer_37366215475450
// MI455X (gfx1250) — matrix operand order repaired and checked by reading, not compiled, not run on hardware
//
#include <hip/hip_runtime.h>
#include <hip/hip_bf16.h>

// ---------------------------------------------------------------------------
// GAT (GraphAttentionLayer) fused implementation for gfx1250 (MI455X).
// B=8, N=1024, INPUT_DIM=256, LAYERS=4, SINGLE=64, ALPHA=0.2
//
// All B-side matrices are kept TRANSPOSED (column-major) in f16 so that one
// lane's 16 B-fragment elements are two 16-byte runs (2 x b128 loads).
//
// Workspace layout:
//   X    : B*N*512 f32   concat buffer. cols [0,256) = gcn_inputs,
//                        cols [256+64l, +64) = AxW of layer l.
//   xW   : B*N*64  f32   per-layer xW (for s_src/s_dst dots)
//   ssrc : B*N     f32   xW @ a_top
//   sdst : B*N     f32   xW @ a_bot
//   mrow : B*N     f32   per-row masked softmax max
//   xWhT : B*64*N  f16   per-layer xW, transposed (TDM source for attention)
//   WhT  : 90112   f16   all layer weights, f16 transposed [64][fan_in]
//   WoutT: 256*256 f16   Wout transposed [256][256]
// Total ~19.4 MiB.
// ---------------------------------------------------------------------------

typedef __attribute__((ext_vector_type(16))) _Float16     v16h;
typedef __attribute__((ext_vector_type(8)))  _Float16     v8h;
typedef __attribute__((ext_vector_type(8)))  float        v8f;
typedef __attribute__((ext_vector_type(4)))  unsigned int v4u;
typedef __attribute__((ext_vector_type(8)))  int          v8i;
typedef __attribute__((ext_vector_type(4)))  int          v4i;

#define BB    8
#define NN    1024
#define DIN   256
#define SGL   64
#define XDIM  512
#define ALPHA 0.2f
#define NEGV  (-9000000000000000.0f)
#define MAXFI 448            // max fan_in (layer 3)

#if defined(__has_builtin)
#if __has_builtin(__builtin_amdgcn_tensor_load_to_lds) && \
    __has_builtin(__builtin_amdgcn_s_wait_tensorcnt)
#define USE_TDM 1
#endif
#endif

__device__ __forceinline__ v8f wmma_f16(v16h a, v16h b, v8f c) {
  // D = A(16x32 f16) * B(32x16 f16) + C(16x16 f32)
  return __builtin_amdgcn_wmma_f32_16x16x32_f16(
      /*neg_a=*/false, a, /*neg_b=*/false, b,
      /*c_mod=*/(short)0, c, /*reuse_a=*/false, /*reuse_b=*/false);
}

// A fragment (16x32, f16) from row-major fp32: two contiguous 8-float runs.
// base = &A[row0 + (lane&15)][k0 + 8*(lane>>4)].
// element e -> K offset (e&7) + ((e>>3)<<4)  (per ISA 16-bit A layout)
__device__ __forceinline__ v16h load_afrag_f32(const float* __restrict__ base) {
  const float4* q0p = (const float4*)base;          // K offsets 0..7
  const float4* q2p = (const float4*)(base + 16);   // K offsets 16..23
  float4 q0 = q0p[0], q1 = q0p[1], q2 = q2p[0], q3 = q2p[1];
  v16h r;
  r[0]=(_Float16)q0.x;  r[1]=(_Float16)q0.y;  r[2]=(_Float16)q0.z;  r[3]=(_Float16)q0.w;
  r[4]=(_Float16)q1.x;  r[5]=(_Float16)q1.y;  r[6]=(_Float16)q1.z;  r[7]=(_Float16)q1.w;
  r[8]=(_Float16)q2.x;  r[9]=(_Float16)q2.y;  r[10]=(_Float16)q2.z; r[11]=(_Float16)q2.w;
  r[12]=(_Float16)q3.x; r[13]=(_Float16)q3.y; r[14]=(_Float16)q3.z; r[15]=(_Float16)q3.w;
  return r;
}

// B fragment (32x16, f16) from a TRANSPOSED f16 matrix (LDS or global):
// rowbase = &Bt[col0 + (lane&15)][k0 + 8*(lane>>4)] -> two 8-half runs,
// i.e. 2 x b128 loads. element e -> K offset (e&7) + ((e>>3)<<4).
__device__ __forceinline__ v16h load_bfrag_T(const _Float16* rowbase) {
  const v8h* p = (const v8h*)rowbase;
  v8h lo = p[0], hi = p[2];
  v16h r;
#pragma unroll
  for (int e = 0; e < 8; ++e) { r[e] = lo[e]; r[8 + e] = hi[e]; }
  return r;
}

__device__ __forceinline__ float lrelu(float x) { return x > 0.f ? x : ALPHA * x; }

#ifdef USE_TDM
// Issue a 2D TDM load of an f16 tile: `width` contiguous elements per row,
// `rows` rows, global row stride `stride` elements; packed into LDS
// (row-major, stride = width).
__device__ __forceinline__ void tdm_load_2d(const _Float16* gsrc, unsigned lds_off,
                                            int width, int rows, int stride) {
  const unsigned long long ga = (unsigned long long)(uintptr_t)gsrc;
  // D# group 0: count=1 | lds_addr | global_addr[56:0] | type=2
  v4u g0 = { 1u, lds_off, (unsigned)(ga & 0xFFFFFFFFu),
             (unsigned)((ga >> 32) & 0x1FFFFFFu) | (2u << 30) };
  // D# group 1: data_size=2B | tensor_dim0=width | tensor_dim1=rows |
  //             tile_dim0=width | tile_dim1=rows | tensor_dim0_stride=stride
  v8i g1 = { (int)0x00010000, (int)((unsigned)width << 16),
             (int)((unsigned)rows << 16), (int)((unsigned)width << 16),
             rows, stride, 0, 0 };
  v4i gz4 = { 0, 0, 0, 0 };
  v8i gz8 = { 0, 0, 0, 0, 0, 0, 0, 0 };
  __builtin_amdgcn_tensor_load_to_lds(g0, g1, gz4, gz4, gz8, 0);
}
#endif

// --------------------------------------------------------------------------
// K-1: convert fp32 [rows x cols] -> f16 TRANSPOSED [cols x rows]
// --------------------------------------------------------------------------
__global__ void gat_cvt_f16_t(const float* __restrict__ src, _Float16* __restrict__ dst,
                              int rows, int cols) {
  const int i = blockIdx.x * blockDim.x + threadIdx.x;
  if (i >= rows * cols) return;
  const int r = i / cols, c = i % cols;
  dst[(size_t)c * rows + r] = (_Float16)src[i];
}

// --------------------------------------------------------------------------
// K0: copy gcn_inputs into X[:, :, 0:256]
// --------------------------------------------------------------------------
__global__ void gat_copy_in(const float* __restrict__ gin, float* __restrict__ X) {
  size_t idx = (size_t)blockIdx.x * blockDim.x + threadIdx.x;
  size_t total = (size_t)BB * NN * DIN;
  if (idx >= total) return;
  size_t row = idx / DIN;
  size_t col = idx % DIN;
  X[row * XDIM + col] = gin[idx];
}

// --------------------------------------------------------------------------
// K1: xW = X[:, :, 0:fan_in] @ W + b       (WMMA GEMM, 16x16x32 f16)
// block = 256 threads (8 waves), wave -> 16 rows x 64 cols, grid (N/128, B).
// The whole transposed f16 W panel [64][fan_in] (<= 56 KB) is DMA'd into LDS
// once by the Tensor Data Mover; B fragments are then 2 x ds_load_b128 each.
// Stores xW as f32 (row-major, for dots) and f16 transposed (for attention).
// --------------------------------------------------------------------------
__global__ void gat_xw_gemm(const float* __restrict__ X,
                            const _Float16* __restrict__ WhT, // [64][fan_in] f16
                            const float* __restrict__ bias,   // 64
                            float* __restrict__ xW,           // B*N*64 f32
                            _Float16* __restrict__ xWhT,      // B*64*N f16
                            int fan_in) {
  __shared__ _Float16 wtile[SGL * MAXFI];      // 56 KB W panel (transposed)

  const int b    = blockIdx.y;
  const int wave = threadIdx.x >> 5;
  const int lane = threadIdx.x & 31;
  const int half = lane >> 4;
  const int ln   = lane & 15;
  const int row0 = blockIdx.x * 128 + wave * 16;
  const float* Xb = X + (size_t)b * NN * XDIM;

#ifdef USE_TDM
  if (wave == 0) {
    tdm_load_2d(WhT, (unsigned)(uintptr_t)(void*)wtile, fan_in, SGL, fan_in);
    __builtin_amdgcn_s_wait_tensorcnt(0);
  }
#else
  {
    const uint4* src = (const uint4*)WhT;
    uint4* dst = (uint4*)wtile;
    for (int t = (int)threadIdx.x; t < fan_in * 8; t += 256)
      dst[t] = src[t];
  }
#endif
  __syncthreads();

  v8f c[4] = {v8f{}, v8f{}, v8f{}, v8f{}};
  for (int k0 = 0; k0 < fan_in; k0 += 32) {
    const float* ap = Xb + (size_t)(row0 + ln) * XDIM + k0 + half * 8;
    v16h a = load_afrag_f32(ap);
#pragma unroll
    for (int t = 0; t < 4; ++t) {
      v16h bf = load_bfrag_T(wtile + (size_t)(t * 16 + ln) * fan_in + k0 + half * 8);
      c[t] = wmma_f16(a, bf, c[t]);
    }
  }
#pragma unroll
  for (int t = 0; t < 4; ++t) {
    const int col = t * 16 + ln;
    const float bv = bias[col];
#pragma unroll
    for (int r = 0; r < 8; ++r) {
      const int row = row0 + half * 8 + r;
      const float v = c[t][r] + bv;
      xW[((size_t)b * NN + row) * SGL + col] = v;
      xWhT[((size_t)b * SGL + col) * NN + row] = (_Float16)v;   // transposed
    }
  }
}

// --------------------------------------------------------------------------
// K2: s_src = xW @ a_top, s_dst = xW @ a_bot  (thread per row)
// --------------------------------------------------------------------------
__global__ void gat_sdot(const float* __restrict__ xW, const float* __restrict__ a,
                         float* __restrict__ ssrc, float* __restrict__ sdst) {
  const int row = blockIdx.x * blockDim.x + threadIdx.x; // 0..B*N-1
  if (row >= BB * NN) return;
  const float* x = xW + (size_t)row * SGL;
  float s1 = 0.f, s2 = 0.f;
#pragma unroll 8
  for (int d = 0; d < SGL; ++d) {
    const float v = x[d];
    s1 += v * a[d];
    s2 += v * a[SGL + d];
  }
  ssrc[row] = s1;
  sdst[row] = s2;
}

// --------------------------------------------------------------------------
// K3: per-row masked softmax max.
// lrelu is monotone => max_j e_ij = lrelu(s_src_i + max over masked j of s_dst_j)
// wave per row, lanes stride j (coalesced adj scan).
// --------------------------------------------------------------------------
__global__ void gat_rowmax(const int* __restrict__ adj, const float* __restrict__ ssrc,
                           const float* __restrict__ sdst, float* __restrict__ mrow) {
  const int wave = threadIdx.x >> 5;
  const int lane = threadIdx.x & 31;
  const int gr   = blockIdx.x * 8 + wave;      // global row in [0, B*N)
  const int b    = gr / NN;
  const int* ar  = adj + (size_t)gr * NN;
  const float* sd = sdst + (size_t)b * NN;

  float mx = -3.4e38f;
  bool any = false;
  for (int j = lane; j < NN; j += 32) {
    if (ar[j] > 0) {
      mx = fmaxf(mx, sd[j]);
      any = true;
    }
  }
#pragma unroll
  for (int off = 16; off > 0; off >>= 1)
    mx = fmaxf(mx, __shfl_xor(mx, off, 32));
  const bool anyw = (__ballot(any) != 0ull);
  if (lane == 0)
    mrow[gr] = anyw ? lrelu(ssrc[gr] + mx) : NEGV;
}

// --------------------------------------------------------------------------
// K4: fused masked softmax + attention @ xW  (flash-style, one pass over j)
//   P_ij = exp((mask ? lrelu(s_i + s_j) : NEG) - m_i)   built as f16 A-frag
//   O    = P @ xW  via WMMA;  l_i = sum_j P_ij accumulated in registers
//   AxW  = relu(O / l)  written into X[:, :, col_off : col_off+64]
//
// Double-buffered TDM pipeline: wave 0 issues the DMA for j-tile t+1 before
// the block computes on tile t (s_wait_tensorcnt after compute -> the Tensor
// Data Mover overlaps the WMMA stream). The tile is the TRANSPOSED xW
// (64 d-rows x 32 j-cols, global row stride N), so B fragments are
// 2 x ds_load_b128. adj/sdst are read as b128 vectors.
// block = 128 threads (4 waves), wave -> 16 rows x all 64 dims. grid (N/64, B)
// --------------------------------------------------------------------------
__global__ void gat_attn(const int* __restrict__ adj, const _Float16* __restrict__ xWhT,
                         const float* __restrict__ ssrc, const float* __restrict__ sdst,
                         const float* __restrict__ mrow, float* __restrict__ X,
                         int col_off) {
  __shared__ _Float16 tile[2][SGL * 32];       // 2 x 4 KB double buffer, [d][jloc]

  const int b    = blockIdx.y;
  const int wave = threadIdx.x >> 5;
  const int lane = threadIdx.x & 31;
  const int half = lane >> 4;
  const int ln   = lane & 15;
  const int i0   = blockIdx.x * 64 + wave * 16;
  const int i    = i0 + ln;                    // row this lane builds P for

  const float si = ssrc[(size_t)b * NN + i];
  const float mi = mrow[(size_t)b * NN + i];
  const int*      adjrow = adj  + ((size_t)b * NN + i) * NN;
  const float*    sd     = sdst + (size_t)b * NN;
  const _Float16* xwhtb  = xWhT + (size_t)b * SGL * NN;   // [64][1024]

  v8f c[4] = {v8f{}, v8f{}, v8f{}, v8f{}};
  float lsum = 0.f;

  // one j-tile of compute: P build + 4 WMMAs against the LDS tile
  auto compute_tile = [&](const _Float16* tl, int j0) {
    if (j0 + 32 < NN)  // pull next adj tile toward the WGP (global_prefetch_b8)
      __builtin_prefetch(adjrow + j0 + 32 + half * 8, 0, 3);

    int   am[16];
    float sv[16];
    {
      const int4*   a1 = (const int4*)(adjrow + j0 + half * 8);
      const int4*   a2 = (const int4*)(adjrow + j0 + 16 + half * 8);
      const float4* s1 = (const float4*)(sd + j0 + half * 8);
      const float4* s2 = (const float4*)(sd + j0 + 16 + half * 8);
      int4 a10 = a1[0], a11 = a1[1], a20 = a2[0], a21 = a2[1];
      float4 s10 = s1[0], s11 = s1[1], s20 = s2[0], s21 = s2[1];
      am[0]=a10.x; am[1]=a10.y; am[2]=a10.z; am[3]=a10.w;
      am[4]=a11.x; am[5]=a11.y; am[6]=a11.z; am[7]=a11.w;
      am[8]=a20.x; am[9]=a20.y; am[10]=a20.z; am[11]=a20.w;
      am[12]=a21.x; am[13]=a21.y; am[14]=a21.z; am[15]=a21.w;
      sv[0]=s10.x; sv[1]=s10.y; sv[2]=s10.z; sv[3]=s10.w;
      sv[4]=s11.x; sv[5]=s11.y; sv[6]=s11.z; sv[7]=s11.w;
      sv[8]=s20.x; sv[9]=s20.y; sv[10]=s20.z; sv[11]=s20.w;
      sv[12]=s21.x; sv[13]=s21.y; sv[14]=s21.z; sv[15]=s21.w;
    }

    v16h p;
#pragma unroll
    for (int e = 0; e < 16; ++e) {
      // A-frag element e covers K = (e&7) + ((e>>3)<<4) + 8*half == am[e]/sv[e]
      const float ev = (am[e] > 0) ? lrelu(si + sv[e]) : NEGV;
      const float pv = __expf(ev - mi);
      lsum += pv;
      p[e] = (_Float16)pv;
    }
#pragma unroll
    for (int t = 0; t < 4; ++t) {
      v16h bf = load_bfrag_T(tl + (size_t)(t * 16 + ln) * 32 + half * 8);
      c[t] = wmma_f16(p, bf, c[t]);
    }
  };

#ifdef USE_TDM
  const unsigned lds0 = (unsigned)(uintptr_t)(void*)&tile[0][0];
  const unsigned lds1 = (unsigned)(uintptr_t)(void*)&tile[1][0];
  if (wave == 0) {
    tdm_load_2d(xwhtb, lds0, 32, SGL, NN);     // prologue: tile 0
    __builtin_amdgcn_s_wait_tensorcnt(0);
  }
  __syncthreads();
  for (int t = 0; t < NN / 32; ++t) {
    // buf[(t+1)&1] was last read in iteration t-1; the barrier at the end of
    // t-1 guarantees those reads completed -> safe to overwrite now.
    if (wave == 0 && t + 1 < NN / 32)
      tdm_load_2d(xwhtb + (size_t)(t + 1) * 32, ((t + 1) & 1) ? lds1 : lds0,
                  32, SGL, NN);
    compute_tile(&tile[t & 1][0], t * 32);
    if (wave == 0 && t + 1 < NN / 32)
      __builtin_amdgcn_s_wait_tensorcnt(0);    // after compute: DMA overlapped
    __syncthreads();
  }
#else
  for (int t = 0; t < NN / 32; ++t) {
    __syncthreads();
    {
      // cooperative strided copy: 64 rows x 32 halfs (64 B = 4 uint4 per row)
      for (int k = (int)threadIdx.x; k < SGL * 4; k += 128) {
        const int d = k >> 2, seg = k & 3;
        const uint4* src = (const uint4*)(xwhtb + (size_t)d * NN + t * 32) + seg;
        ((uint4*)&tile[0][0])[k] = *src;
      }
    }
    __syncthreads();
    compute_tile(&tile[0][0], t * 32);
  }
#endif

  // lanes L and L^16 both built P for row (L&15); combine their partial sums.
  lsum += __shfl_xor(lsum, 16, 32);   // lane L now holds l for row i0 + (L&15)

#pragma unroll
  for (int t = 0; t < 4; ++t) {
    const int col = col_off + t * 16 + ln;
#pragma unroll
    for (int r = 0; r < 8; ++r) {
      const int row = i0 + half * 8 + r;
      const float l = __shfl(lsum, half * 8 + r, 32); // row-stat lives in lane (row&15)
      const float v = c[t][r] / l;
      X[((size_t)b * NN + row) * XDIM + col] = v > 0.f ? v : 0.f;
    }
  }
}

// --------------------------------------------------------------------------
// K5: out = (X[:, :, 256:512] + gcn_inputs) @ Wout + bout
// block = 256 threads (8 waves); wave -> (row-tile, col-tile): 32 rows x 256 cols.
// B fragments come from transposed f16 Wout: 2 x global_load_b128 each.
// --------------------------------------------------------------------------
__global__ void gat_out_gemm(const float* __restrict__ X, const float* __restrict__ gin,
                             const _Float16* __restrict__ WoutT, // [256][256] f16
                             const float* __restrict__ bout,
                             float* __restrict__ out) {
  const int wave = threadIdx.x >> 5;
  const int rt   = wave >> 2;        // 0..1
  const int ct   = wave & 3;         // 0..3
  const int lane = threadIdx.x & 31;
  const int half = lane >> 4;
  const int ln   = lane & 15;
  const int gr0  = blockIdx.x * 32 + rt * 16;   // global row over B*N

  v8f c[4] = {v8f{}, v8f{}, v8f{}, v8f{}};
  for (int k0 = 0; k0 < DIN; k0 += 32) {
    // A = X[:, 256+k] + gin[:, k]  (residual fused into vectorized frag load)
    const int arow = gr0 + ln;
    const float* xp = X   + (size_t)arow * XDIM + DIN + k0 + half * 8;
    const float* gp = gin + (size_t)arow * DIN  +       k0 + half * 8;
    v16h a;
    {
      const float4* x0 = (const float4*)xp;        // K 0..7
      const float4* x2 = (const float4*)(xp + 16); // K 16..23
      const float4* g0 = (const float4*)gp;
      const float4* g2 = (const float4*)(gp + 16);
      float4 q0 = x0[0], q1 = x0[1], q2 = x2[0], q3 = x2[1];
      float4 p0 = g0[0], p1 = g0[1], p2 = g2[0], p3 = g2[1];
      a[0]=(_Float16)(q0.x+p0.x);  a[1]=(_Float16)(q0.y+p0.y);
      a[2]=(_Float16)(q0.z+p0.z);  a[3]=(_Float16)(q0.w+p0.w);
      a[4]=(_Float16)(q1.x+p1.x);  a[5]=(_Float16)(q1.y+p1.y);
      a[6]=(_Float16)(q1.z+p1.z);  a[7]=(_Float16)(q1.w+p1.w);
      a[8]=(_Float16)(q2.x+p2.x);  a[9]=(_Float16)(q2.y+p2.y);
      a[10]=(_Float16)(q2.z+p2.z); a[11]=(_Float16)(q2.w+p2.w);
      a[12]=(_Float16)(q3.x+p3.x); a[13]=(_Float16)(q3.y+p3.y);
      a[14]=(_Float16)(q3.z+p3.z); a[15]=(_Float16)(q3.w+p3.w);
    }
#pragma unroll
    for (int t = 0; t < 4; ++t) {
      const int col = ct * 64 + t * 16 + ln;
      v16h bf = load_bfrag_T(WoutT + (size_t)col * DIN + k0 + half * 8);
      c[t] = wmma_f16(a, bf, c[t]);
    }
  }
#pragma unroll
  for (int t = 0; t < 4; ++t) {
    const int col = ct * 64 + t * 16 + ln;
    const float bv = bout[col];
#pragma unroll
    for (int r = 0; r < 8; ++r) {
      const int row = gr0 + half * 8 + r;
      out[(size_t)row * DIN + col] = c[t][r] + bv;
    }
  }
}

// --------------------------------------------------------------------------
extern "C" void kernel_launch(void* const* d_in, const int* in_sizes, int n_in,
                              void* d_out, int out_size, void* d_ws, size_t ws_size,
                              hipStream_t stream) {
  const float* gin = (const float*)d_in[0];
  const int*   adj = (const int*)d_in[1];

  const float* Wl[4];
  const float* bl[4];
  const float *a, *Wout, *bout;
  if (n_in >= 13) {               // tuples flattened to individual entries
    for (int l = 0; l < 4; ++l) Wl[l] = (const float*)d_in[2 + l];
    for (int l = 0; l < 4; ++l) bl[l] = (const float*)d_in[6 + l];
    a    = (const float*)d_in[10];
    Wout = (const float*)d_in[11];
    bout = (const float*)d_in[12];
  } else {                        // tuples passed as concatenated blobs
    const float* wbuf = (const float*)d_in[2];
    size_t off = 0;
    for (int l = 0; l < 4; ++l) { Wl[l] = wbuf + off; off += (size_t)(DIN + SGL * l) * SGL; }
    const float* bbuf = (const float*)d_in[3];
    for (int l = 0; l < 4; ++l) bl[l] = bbuf + (size_t)SGL * l;
    a    = (const float*)d_in[4];
    Wout = (const float*)d_in[5];
    bout = (const float*)d_in[6];
  }
  float* out = (float*)d_out;

  // workspace carve-up
  float*    X    = (float*)d_ws;                        // B*N*512 f32
  float*    xW   = X    + (size_t)BB * NN * XDIM;       // B*N*64 f32
  float*    ssrc = xW   + (size_t)BB * NN * SGL;        // B*N
  float*    sdst = ssrc + (size_t)BB * NN;              // B*N
  float*    mrow = sdst + (size_t)BB * NN;              // B*N
  _Float16* xWhT = (_Float16*)(mrow + (size_t)BB * NN); // B*64*N f16 (transposed)
  _Float16* WhT  = xWhT + (size_t)BB * SGL * NN;        // 90112 f16 (all layers, T)
  _Float16* WoutT = WhT + 90112;                        // 256*256 f16 (T)

  // per-layer offsets into WhT
  size_t woff[4];
  {
    size_t o = 0;
    for (int l = 0; l < 4; ++l) { woff[l] = o; o += (size_t)(DIN + SGL * l) * SGL; }
  }

  // prep: transposed f16 copies of weights (tiny)
  for (int l = 0; l < 4; ++l) {
    const int fi = DIN + SGL * l;
    gat_cvt_f16_t<<<(fi * SGL + 255) / 256, 256, 0, stream>>>(Wl[l], WhT + woff[l],
                                                              fi, SGL);
  }
  gat_cvt_f16_t<<<(DIN * DIN + 255) / 256, 256, 0, stream>>>(Wout, WoutT, DIN, DIN);

  // K0: seed X with gcn_inputs
  {
    const size_t total = (size_t)BB * NN * DIN;
    gat_copy_in<<<(unsigned)((total + 255) / 256), 256, 0, stream>>>(gin, X);
  }

  for (int l = 0; l < 4; ++l) {
    const int fan_in = DIN + SGL * l;
    dim3 g1(NN / 128, BB);
    gat_xw_gemm<<<g1, 256, 0, stream>>>(X, WhT + woff[l], bl[l], xW, xWhT, fan_in);

    gat_sdot<<<(BB * NN) / 256, 256, 0, stream>>>(xW, a, ssrc, sdst);

    gat_rowmax<<<(BB * NN) / 8, 256, 0, stream>>>(adj, ssrc, sdst, mrow);

    dim3 g4(NN / 64, BB);
    gat_attn<<<g4, 128, 0, stream>>>(adj, xWhT, ssrc, sdst, mrow, X, DIN + SGL * l);
  }

  gat_out_gemm<<<(BB * NN) / 32, 256, 0, stream>>>(X, gin, WoutT, bout, out);
}
